// LinearAttention_15745350107530
// MI455X (gfx1250) — matrix operand order repaired and checked by reading, not compiled, not run on hardware
//
#include <hip/hip_runtime.h>
#include <hip/hip_bf16.h>

// LinearAttention on MI455X (gfx1250): wave32, WMMA bf16 16x16x32 (f32 acc),
// double-buffered LDS tiles, async global->LDS copies where operands are bf16.
// B=8, DIM=512, N=4096, HEADS=8, DIM_HEAD=64, HID=512.

#define N_TOK 4096
#define DIMC  512
#define STR   72   // LDS row stride in ushorts (144B -> every frag half 16B aligned)

#if __has_builtin(__builtin_amdgcn_global_load_async_to_lds_b128) && \
    __has_builtin(__builtin_amdgcn_s_wait_asynccnt)
#define ASYNC_OK 1
#else
#define ASYNC_OK 0
#endif

#if ASYNC_OK
#define ASYNC_WAIT(n) __builtin_amdgcn_s_wait_asynccnt(n)
#else
#define ASYNC_WAIT(n)
#endif

typedef __attribute__((ext_vector_type(16))) __bf16 v16bf;
typedef __attribute__((ext_vector_type(8)))  __bf16 v8bf;
typedef __attribute__((ext_vector_type(8)))  float  v8f;
typedef __attribute__((ext_vector_type(4)))  int    v4i;

__device__ __forceinline__ unsigned short f32_bf16(float f) {
  unsigned int u = __float_as_uint(f);
  u += 0x7FFFu + ((u >> 16) & 1u);          // round-to-nearest-even
  return (unsigned short)(u >> 16);
}
__device__ __forceinline__ float bf16_f32(unsigned short u) {
  return __uint_as_float(((unsigned int)u) << 16);
}
__device__ __forceinline__ unsigned int pack2(float lo, float hi) {
  return (unsigned int)f32_bf16(lo) | ((unsigned int)f32_bf16(hi) << 16);
}
__device__ __forceinline__ v16bf cat8(v8bf lo, v8bf hi) {
  return __builtin_shufflevector(lo, hi, 0,1,2,3,4,5,6,7,8,9,10,11,12,13,14,15);
}
// A fragment: lanes 0-15 K {0..7,16..23}; lanes 16-31 K {8..15,24..31}
__device__ __forceinline__ v16bf fragA(const unsigned short* rowp, int hi) {
  v8bf l = *(const v8bf*)(rowp + (hi ? 8  : 0));
  v8bf h = *(const v8bf*)(rowp + (hi ? 24 : 16));
  return cat8(l, h);
}
// B fragment: lanes 0-15 K {0..7,16..23}; lanes 16-31 K {8..15,24..31}
__device__ __forceinline__ v16bf fragB(const unsigned short* colp, int hi) {
  v8bf l = *(const v8bf*)(colp + (hi ? 8 : 0));
  v8bf h = *(const v8bf*)(colp + (hi ? 24 : 16));
  return cat8(l, h);
}
__device__ __forceinline__ v8f wmma_bf16(v16bf a, v16bf b, v8f c) {
  return __builtin_amdgcn_wmma_f32_16x16x32_bf16(false, a, false, b, (short)0, c,
                                                 false, false);
}
// 16-byte global->LDS copy (async engine when available, ASYNCcnt-tracked)
__device__ __forceinline__ void cp16(unsigned short* l, const unsigned short* g) {
#if ASYNC_OK
  __builtin_amdgcn_global_load_async_to_lds_b128(
      (__attribute__((address_space(1))) v4i*)(g),
      (__attribute__((address_space(3))) v4i*)(l), 0, 0);
#else
  *(uint4*)l = *(const uint4*)g;
#endif
}

// ---- staging helpers (256 threads) ------------------------------------------------
// A tile 128x64: f32 row-major (lda) -> bf16 LDS [m][k], b128 loads + b128 stores
__device__ __forceinline__ void stageA_f32(unsigned short* As, const float* A,
                                           int lda, int k0, int tid) {
#pragma unroll
  for (int i = 0; i < 2; ++i) {
    int m = (tid >> 2) + i * 64;
    int kc = (tid & 3) * 16;
    const float4* src = (const float4*)(A + (size_t)m * lda + k0 + kc);
    unsigned int pk[8];
#pragma unroll
    for (int j = 0; j < 4; ++j) {
      float4 f = src[j];
      pk[2 * j]     = pack2(f.x, f.y);
      pk[2 * j + 1] = pack2(f.z, f.w);
    }
    uint4* dst = (uint4*)&As[m * STR + kc];
    dst[0] = make_uint4(pk[0], pk[1], pk[2], pk[3]);
    dst[1] = make_uint4(pk[4], pk[5], pk[6], pk[7]);
  }
}
// B tile 64(K)x128(N): f32 [k][n] (ldb) -> bf16 LDS [n][k] (transpose), pair-packed
__device__ __forceinline__ void stageB_f32(unsigned short* Bs, const float* Bp,
                                           int ldb, int k0, int tid) {
#pragma unroll
  for (int c = 0; c < 4; ++c) {
    int id = tid + c * 256;
    int kp = id >> 5;                    // 0..31 -> k pair 2*kp
    int nq = (id & 31) * 4;
    const float4 f0 = *(const float4*)(Bp + (size_t)(k0 + 2 * kp) * ldb + nq);
    const float4 f1 = *(const float4*)(Bp + (size_t)(k0 + 2 * kp + 1) * ldb + nq);
    *(unsigned int*)&Bs[(nq + 0) * STR + 2 * kp] = pack2(f0.x, f1.x);
    *(unsigned int*)&Bs[(nq + 1) * STR + 2 * kp] = pack2(f0.y, f1.y);
    *(unsigned int*)&Bs[(nq + 2) * STR + 2 * kp] = pack2(f0.z, f1.z);
    *(unsigned int*)&Bs[(nq + 3) * STR + 2 * kp] = pack2(f0.w, f1.w);
  }
}
// B tile 64(K)x128(N): bf16 [k][n] (ldb) -> bf16 LDS [n][k] (transpose)
__device__ __forceinline__ void stageB_bf16(unsigned short* Bs,
                                            const unsigned short* Bp, int ldb,
                                            int k0, int tid) {
#pragma unroll
  for (int c = 0; c < 4; ++c) {
    int id = tid + c * 256;
    int kp = id >> 5;
    int nq = (id & 31) * 4;
    const uint2 a = *(const uint2*)(Bp + (size_t)(k0 + 2 * kp) * ldb + nq);
    const uint2 b = *(const uint2*)(Bp + (size_t)(k0 + 2 * kp + 1) * ldb + nq);
    *(unsigned int*)&Bs[(nq + 0) * STR + 2 * kp] = (a.x & 0xFFFFu) | (b.x << 16);
    *(unsigned int*)&Bs[(nq + 1) * STR + 2 * kp] = (a.x >> 16) | (b.x & 0xFFFF0000u);
    *(unsigned int*)&Bs[(nq + 2) * STR + 2 * kp] = (a.y & 0xFFFFu) | (b.y << 16);
    *(unsigned int*)&Bs[(nq + 3) * STR + 2 * kp] = (a.y >> 16) | (b.y & 0xFFFF0000u);
  }
}
// 64x64 bf16 tile copy, K-contiguous source (row stride lds elements)
__device__ __forceinline__ void stage_copy64(unsigned short* dst,
                                             const unsigned short* src, int lds,
                                             int k0, int tid) {
#pragma unroll
  for (int c = 0; c < 2; ++c) {
    int id = tid + c * 256;
    int r = id >> 3, off = (id & 7) * 8;
    cp16(&dst[r * STR + off], &src[(size_t)r * lds + k0 + off]);
  }
}

// ---------------- Kernel 1: qkv = w_qkv(1536x512) @ x[b](512x4096) -----------------
// Block tile 128x128, K-step 64, double-buffered. Writes q,k,v bf16 (Bx512xN each).
__global__ __launch_bounds__(256) void k1_qkv(
    const float* __restrict__ w_qkv, const float* __restrict__ x,
    unsigned short* __restrict__ qb, unsigned short* __restrict__ kb,
    unsigned short* __restrict__ vb) {
  const int tid = threadIdx.x, lane = tid & 31, wv = tid >> 5;
  const int nt = blockIdx.x, mt = blockIdx.y, b = blockIdx.z;
  __shared__ __align__(16) unsigned short As[2][128 * STR];
  __shared__ __align__(16) unsigned short Bs[2][128 * STR];
  const float* A  = w_qkv + (size_t)(mt * 128) * DIMC;
  const float* Bp = x + (size_t)b * DIMC * N_TOK + nt * 128;
  v8f acc[2][4] = {};
  const int wm = wv & 3, wn = wv >> 2;
  const int hi = lane >> 4, l15 = lane & 15;

  stageA_f32(As[0], A, DIMC, 0, tid);
  stageB_f32(Bs[0], Bp, N_TOK, 0, tid);
  __syncthreads();
  int buf = 0;
  for (int k0 = 0; k0 < DIMC; k0 += 64) {
    if (k0 + 64 < DIMC) {
      stageA_f32(As[buf ^ 1], A, DIMC, k0 + 64, tid);
      stageB_f32(Bs[buf ^ 1], Bp, N_TOK, k0 + 64, tid);
    }
#pragma unroll
    for (int kf = 0; kf < 2; ++kf) {
      v16bf af[2];
#pragma unroll
      for (int i = 0; i < 2; ++i)
        af[i] = fragA(&As[buf][(wm * 32 + i * 16 + l15) * STR + kf * 32], hi);
#pragma unroll
      for (int j = 0; j < 4; ++j) {
        v16bf bf = fragB(&Bs[buf][(wn * 64 + j * 16 + l15) * STR + kf * 32], hi);
#pragma unroll
        for (int i = 0; i < 2; ++i) acc[i][j] = wmma_bf16(af[i], bf, acc[i][j]);
      }
    }
    __syncthreads();
    buf ^= 1;
  }
#pragma unroll
  for (int i = 0; i < 2; ++i)
#pragma unroll
    for (int j = 0; j < 4; ++j)
#pragma unroll
      for (int r = 0; r < 8; ++r) {
        int o = mt * 128 + wm * 32 + i * 16 + r + hi * 8;
        int n = nt * 128 + wn * 64 + j * 16 + l15;
        int sect = o >> 9, oc = o & 511;       // 0:q 1:k 2:v (tile-uniform)
        size_t dst = ((size_t)b * DIMC + oc) * N_TOK + n;
        float vval = acc[i][j][r];
        if      (sect == 0) qb[dst] = f32_bf16(vval);
        else if (sect == 1) kb[dst] = f32_bf16(vval);
        else                vb[dst] = f32_bf16(vval);
      }
}

// ---------------- Kernel 2: in-place softmax over N for each of B*512 k-rows -------
__global__ __launch_bounds__(256) void k2_softmax(unsigned short* __restrict__ kb) {
  const int tid = threadIdx.x;
  unsigned short* p = kb + (size_t)blockIdx.x * N_TOK;
  __shared__ float red[8];
  float mx = -3.402823466e38f;
  for (int i = tid; i < N_TOK; i += 256) mx = fmaxf(mx, bf16_f32(p[i]));
#pragma unroll
  for (int off = 16; off > 0; off >>= 1) mx = fmaxf(mx, __shfl_xor(mx, off, 32));
  if ((tid & 31) == 0) red[tid >> 5] = mx;
  __syncthreads();
  mx = red[0];
#pragma unroll
  for (int i = 1; i < 8; ++i) mx = fmaxf(mx, red[i]);
  __syncthreads();
  float s = 0.f;
  for (int i = tid; i < N_TOK; i += 256) s += __expf(bf16_f32(p[i]) - mx);
#pragma unroll
  for (int off = 16; off > 0; off >>= 1) s += __shfl_xor(s, off, 32);
  if ((tid & 31) == 0) red[tid >> 5] = s;
  __syncthreads();
  s = 0.f;
#pragma unroll
  for (int i = 0; i < 8; ++i) s += red[i];
  const float inv = 1.0f / s;
  for (int i = tid; i < N_TOK; i += 256)
    p[i] = f32_bf16(__expf(bf16_f32(p[i]) - mx) * inv);
}

// ---------------- Kernel 3: context[d,e] = sum_n k[d,n]*v[e,n], per (b,h) ----------
// Async double-buffered 64x64 operand tiles over the 4096-deep reduction.
__global__ __launch_bounds__(256) void k3_context(
    const unsigned short* __restrict__ kb, const unsigned short* __restrict__ vb,
    unsigned short* __restrict__ ctxT) {
  const int tid = threadIdx.x, lane = tid & 31, wv = tid >> 5;
  const int bh = blockIdx.x;
  const unsigned short* kp = kb + (size_t)bh * 64 * N_TOK;
  const unsigned short* vp = vb + (size_t)bh * 64 * N_TOK;
  __shared__ __align__(16) unsigned short As[2][64 * STR];
  __shared__ __align__(16) unsigned short Bs[2][64 * STR];
  v8f acc[2] = {};
  const int wm = wv & 3, wn = wv >> 2;
  const int hi = lane >> 4, l15 = lane & 15;

  stage_copy64(As[0], kp, N_TOK, 0, tid);
  stage_copy64(Bs[0], vp, N_TOK, 0, tid);
  int buf = 0;
  for (int k0 = 0; k0 < N_TOK; k0 += 64) {
    const bool hn = (k0 + 64) < N_TOK;
    if (hn) {
      stage_copy64(As[buf ^ 1], kp, N_TOK, k0 + 64, tid);
      stage_copy64(Bs[buf ^ 1], vp, N_TOK, k0 + 64, tid);
    }
    if (hn) { ASYNC_WAIT(4); } else { ASYNC_WAIT(0); }
    __syncthreads();                       // stage(buf) visible to all waves
#pragma unroll
    for (int kf = 0; kf < 2; ++kf) {
      v16bf a = fragA(&As[buf][(wm * 16 + l15) * STR + kf * 32], hi);
#pragma unroll
      for (int j = 0; j < 2; ++j) {
        v16bf bf = fragB(&Bs[buf][(wn * 32 + j * 16 + l15) * STR + kf * 32], hi);
        acc[j] = wmma_bf16(a, bf, acc[j]);
      }
    }
    __syncthreads();                       // done reading buf before it's re-staged
    buf ^= 1;
  }
  unsigned short* ct = ctxT + (size_t)bh * 64 * 64;
#pragma unroll
  for (int j = 0; j < 2; ++j)
#pragma unroll
    for (int r = 0; r < 8; ++r) {
      int d = wm * 16 + r + hi * 8;
      int e = wn * 32 + j * 16 + l15;
      ct[e * 64 + d] = f32_bf16(acc[j][r]);   // store transposed
    }
}

// ---------------- Kernel 4: out[e,n] = sum_d ctxT[e,d]*q[d,n], per (b,h) -----------
__global__ __launch_bounds__(256) void k4_attn_out(
    const unsigned short* __restrict__ ctxT, const unsigned short* __restrict__ qb,
    unsigned short* __restrict__ outb) {
  const int tid = threadIdx.x, lane = tid & 31, wv = tid >> 5;
  const int nt = blockIdx.x, bh = blockIdx.y;
  const int n0 = nt * 128;
  const unsigned short* ct = ctxT + (size_t)bh * 64 * 64;
  const unsigned short* qp = qb + (size_t)bh * 64 * N_TOK + n0;
  __shared__ __align__(16) unsigned short As[64 * STR];
  __shared__ __align__(16) unsigned short Bs[128 * STR];
  v8f acc[4] = {};
  const int wm = wv & 3, wn = wv >> 2;
  const int hi = lane >> 4, l15 = lane & 15;

  stage_copy64(As, ct, 64, 0, tid);       // ctxT 64x64, async copy
  stageB_bf16(Bs, qp, N_TOK, 0, tid);     // q tile 64x128, transpose
  ASYNC_WAIT(0);
  __syncthreads();
#pragma unroll
  for (int kf = 0; kf < 2; ++kf) {
    v16bf a = fragA(&As[(wm * 16 + l15) * STR + kf * 32], hi);
#pragma unroll
    for (int j = 0; j < 4; ++j) {
      v16bf bf = fragB(&Bs[(wn * 64 + j * 16 + l15) * STR + kf * 32], hi);
      acc[j] = wmma_bf16(a, bf, acc[j]);
    }
  }
#pragma unroll
  for (int j = 0; j < 4; ++j)
#pragma unroll
    for (int r = 0; r < 8; ++r) {
      int e = wm * 16 + r + hi * 8;
      int n = n0 + wn * 64 + j * 16 + l15;
      outb[((size_t)bh * 64 + e) * N_TOK + n] = f32_bf16(acc[j][r]);
    }
}

// ---------------- Kernel 5: y = w_out(512x512) @ out(512x4096) + b_out -------------
__global__ __launch_bounds__(256) void k5_final(
    const float* __restrict__ w_out, const unsigned short* __restrict__ outb,
    const float* __restrict__ b_out, float* __restrict__ y) {
  const int tid = threadIdx.x, lane = tid & 31, wv = tid >> 5;
  const int nt = blockIdx.x, mt = blockIdx.y, b = blockIdx.z;
  __shared__ __align__(16) unsigned short As[2][128 * STR];
  __shared__ __align__(16) unsigned short Bs[2][128 * STR];
  const float* A = w_out + (size_t)(mt * 128) * DIMC;
  const unsigned short* Bp = outb + (size_t)b * DIMC * N_TOK + nt * 128;
  v8f acc[2][4] = {};
  const int wm = wv & 3, wn = wv >> 2;
  const int hi = lane >> 4, l15 = lane & 15;

  stageA_f32(As[0], A, DIMC, 0, tid);
  stageB_bf16(Bs[0], Bp, N_TOK, 0, tid);
  __syncthreads();
  int buf = 0;
  for (int k0 = 0; k0 < DIMC; k0 += 64) {
    if (k0 + 64 < DIMC) {
      stageA_f32(As[buf ^ 1], A, DIMC, k0 + 64, tid);
      stageB_bf16(Bs[buf ^ 1], Bp, N_TOK, k0 + 64, tid);
    }
#pragma unroll
    for (int kf = 0; kf < 2; ++kf) {
      v16bf af[2];
#pragma unroll
      for (int i = 0; i < 2; ++i)
        af[i] = fragA(&As[buf][(wm * 32 + i * 16 + l15) * STR + kf * 32], hi);
#pragma unroll
      for (int j = 0; j < 4; ++j) {
        v16bf bf = fragB(&Bs[buf][(wn * 64 + j * 16 + l15) * STR + kf * 32], hi);
#pragma unroll
        for (int i = 0; i < 2; ++i) acc[i][j] = wmma_bf16(af[i], bf, acc[i][j]);
      }
    }
    __syncthreads();
    buf ^= 1;
  }
#pragma unroll
  for (int i = 0; i < 2; ++i)
#pragma unroll
    for (int j = 0; j < 4; ++j)
#pragma unroll
      for (int r = 0; r < 8; ++r) {
        int o = mt * 128 + wm * 32 + i * 16 + r + hi * 8;
        int n = nt * 128 + wn * 64 + j * 16 + l15;
        y[((size_t)b * DIMC + o) * N_TOK + n] = acc[i][j][r] + b_out[o];
      }
}

extern "C" void kernel_launch(void* const* d_in, const int* in_sizes, int n_in,
                              void* d_out, int out_size, void* d_ws, size_t ws_size,
                              hipStream_t stream) {
  const float* x     = (const float*)d_in[0];   // (8,512,4096)
  const float* w_qkv = (const float*)d_in[1];   // (1536,512)
  const float* w_out = (const float*)d_in[2];   // (512,512)
  const float* b_out = (const float*)d_in[3];   // (512,)
  float* y = (float*)d_out;                     // (8,512,4096)

  char* ws = (char*)d_ws;
  const size_t SZ = (size_t)8 * DIMC * N_TOK;   // elements of a (B,512,N) tensor
  unsigned short* qb   = (unsigned short*)(ws);
  unsigned short* kb   = (unsigned short*)(ws + 2 * SZ);
  unsigned short* vb   = (unsigned short*)(ws + 4 * SZ);
  unsigned short* ctxT = (unsigned short*)(ws + 6 * SZ);
  unsigned short* outb = (unsigned short*)(ws + 6 * SZ + (size_t)64 * 64 * 64 * 2);

  dim3 blk(256, 1, 1);
  k1_qkv    <<<dim3(32, 12, 8), blk, 0, stream>>>(w_qkv, x, qb, kb, vb);
  k2_softmax<<<dim3(4096, 1, 1), blk, 0, stream>>>(kb);
  k3_context<<<dim3(64, 1, 1),  blk, 0, stream>>>(kb, vb, ctxT);
  k4_attn_out<<<dim3(32, 64, 1), blk, 0, stream>>>(ctxT, qb, outb);
  k5_final  <<<dim3(32, 4, 8),  blk, 0, stream>>>(w_out, outb, b_out, y);
}
